// NewsEncoder_10316511445609
// MI455X (gfx1250) — matrix operand order repaired and checked by reading, not compiled, not run on hardware
//
#include <hip/hip_runtime.h>
#include <hip/hip_bf16.h>

typedef __attribute__((ext_vector_type(16))) __bf16 v16bf;
typedef __attribute__((ext_vector_type(8)))  __bf16 bf16x8;
typedef __attribute__((ext_vector_type(4)))  __bf16 bf16x4;
typedef __attribute__((ext_vector_type(8)))  float  v8f;

#define N_ROWS 8192
#define A_ART  32
#define D_DIM  768
#define O_DIM  256
#define LN_EPS 1e-5f

// ---------------------------------------------------------------------------
// Kernel 0: pack proj_w [768,256] f32 row-major -> bf16 in this code's per-lane
// WMMA B-operand order (contiguous k, not AMD's table order; B 32x16: lanes 0-15 hold K=0..15 of col N=lane,
// lanes 16-31 hold K=16..31; 2 K-values per VGPR -> element j of a v16bf is K=j).
// Layout: bw[((kTile*16 + nTile)*32 + lane)*16 + j]
// ---------------------------------------------------------------------------
__global__ void pack_projw_kernel(const float* __restrict__ pw,
                                  __bf16* __restrict__ bw) {
    int idx = blockIdx.x * 256 + threadIdx.x;   // 0 .. 768*256-1
    int k = idx >> 8;          // 0..767  (row of proj_w = K)
    int n = idx & 255;         // 0..255  (col of proj_w = N)
    int kTile = k >> 5, kIn = k & 31;
    int nTile = n >> 4, nIn = n & 15;
    int lane = ((kIn >> 4) << 4) | nIn;   // laneHi*16 + nIn
    int j = kIn & 15;
    size_t dst = (((size_t)(kTile * 16 + nTile)) * 32 + lane) * 16 + j;
    bw[dst] = (__bf16)pw[idx];
}

// ---------------------------------------------------------------------------
// Kernel 1: fused attention-pool. One workgroup (256 thr = 8 waves) per row n.
// Single pass over articles[n] (96 KB f32 -> 48 KB bf16 staged in LDS).
// ---------------------------------------------------------------------------
__global__ void attn_pool_kernel(const float* __restrict__ articles,
                                 const int*   __restrict__ counts,
                                 const float* __restrict__ attn_w,
                                 const float* __restrict__ attn_b,
                                 __bf16* __restrict__ poolBf) {
    __shared__ __bf16 ldsArt[A_ART * D_DIM];   // 48 KB
    __shared__ float  attnS[D_DIM];            // 3 KB
    __shared__ float  sS[A_ART];

    const int n  = blockIdx.x;
    const int t  = threadIdx.x;
    const int wv = t >> 5;
    const int lane = t & 31;

    // stage attn_w
    for (int i = t; i < D_DIM; i += 256) attnS[i] = attn_w[i];

    // stage articles[n] as bf16 (float4-coalesced: 6144 float4s / 256 threads)
    const float4* g4 = (const float4*)(articles + (size_t)n * A_ART * D_DIM);
    bf16x4* l4 = (bf16x4*)ldsArt;
    #pragma unroll
    for (int i = 0; i < 24; ++i) {
        float4 f = g4[t + i * 256];
        bf16x4 b;
        b[0] = (__bf16)f.x; b[1] = (__bf16)f.y;
        b[2] = (__bf16)f.z; b[3] = (__bf16)f.w;
        l4[t + i * 256] = b;
    }
    __syncthreads();

    // scores: wave w handles articles 4w..4w+3; lane-strided dot over 768
    #pragma unroll
    for (int q = 0; q < 4; ++q) {
        const int a = wv * 4 + q;
        float p = 0.f;
        #pragma unroll
        for (int j = 0; j < 24; ++j) {
            const int d = lane + j * 32;
            p += (float)ldsArt[a * D_DIM + d] * attnS[d];
        }
        #pragma unroll
        for (int off = 16; off; off >>= 1) p += __shfl_xor(p, off, 32);
        if (lane == 0) sS[a] = p + attn_b[0];
    }
    __syncthreads();

    // masked softmax, computed redundantly per thread (deterministic, cheap)
    const int cnt = counts[n];
    float e[A_ART];
    float m = -3.0e38f;
    #pragma unroll
    for (int a = 0; a < A_ART; ++a) {
        float s = (a < cnt) ? sS[a] : -1e9f;
        e[a] = s;
        m = fmaxf(m, s);
    }
    float sum = 0.f;
    #pragma unroll
    for (int a = 0; a < A_ART; ++a) {
        float x = __expf(e[a] - m);
        e[a] = x;
        sum += x;
    }
    const float inv = 1.f / sum;

    // weighted pool: each thread owns 3 d indices, d = t + 256k
    #pragma unroll
    for (int k = 0; k < 3; ++k) {
        const int d = t + k * 256;
        float p = 0.f;
        #pragma unroll
        for (int a = 0; a < A_ART; ++a)
            p += e[a] * (float)ldsArt[a * D_DIM + d];
        poolBf[(size_t)n * D_DIM + d] = (__bf16)(p * inv);
    }
}

// ---------------------------------------------------------------------------
// Kernel 2: WMMA GEMM [16 x 768] x [768 x 256] + bias + LayerNorm + exact GELU
// + count==0 fallback, fused per 16-row block. 8 waves x 2 col-tiles each.
// ---------------------------------------------------------------------------
__global__ void gemm_ln_gelu_kernel(const __bf16* __restrict__ poolBf,
                                    const __bf16* __restrict__ bw,
                                    const float*  __restrict__ proj_b,
                                    const float*  __restrict__ ln_w,
                                    const float*  __restrict__ ln_b,
                                    const int*    __restrict__ counts,
                                    const float*  __restrict__ no_news,
                                    float* __restrict__ out) {
    __shared__ float hS[16 * O_DIM];   // 16 KB

    const int t      = threadIdx.x;
    const int wv     = t >> 5;
    const int lane   = t & 31;
    const int laneLo = lane & 15;
    const int laneHi = lane >> 4;
    const int rBase  = blockIdx.x * 16;

    const __bf16* aRow = poolBf + (size_t)(rBase + laneLo) * D_DIM;

    #pragma unroll
    for (int ci = 0; ci < 2; ++ci) {
        const int ct = wv * 2 + ci;            // col tile 0..15
        v8f c = {0.f, 0.f, 0.f, 0.f, 0.f, 0.f, 0.f, 0.f};

        for (int kt = 0; kt < 24; ++kt) {
            // A operand: lane holds row M=laneLo,
            // element j = K[laneHi*16 + j] (matches packed B order)
            union { v16bf v; bf16x8 h[2]; } au;
            au.h[0] = *(const bf16x8*)(aRow + kt * 32 + laneHi * 16);
            au.h[1] = *(const bf16x8*)(aRow + kt * 32 + laneHi * 16 + 8);
            // B operand: pre-swizzled, one contiguous 32B load per lane
            v16bf b = *(const v16bf*)(bw + (((size_t)kt * 16 + ct) * 32 + lane) * 16);
            c = __builtin_amdgcn_wmma_f32_16x16x32_bf16(
                    false, au.v, false, b, (short)0, c, false, false);
        }

        // bias + scatter C tile to LDS (C layout: VGPR r -> M=r+8*laneHi, N=laneLo)
        const float bias = proj_b[ct * 16 + laneLo];
        #pragma unroll
        for (int r = 0; r < 8; ++r) {
            hS[(r + 8 * laneHi) * O_DIM + ct * 16 + laneLo] = c[r] + bias;
        }
    }
    __syncthreads();

    // LayerNorm + GELU + fallback: wave w handles rows 2w, 2w+1
    #pragma unroll
    for (int ri = 0; ri < 2; ++ri) {
        const int row = wv * 2 + ri;
        const int n   = rBase + row;
        float x[8], s = 0.f, ss = 0.f;
        #pragma unroll
        for (int j = 0; j < 8; ++j) {
            float v = hS[row * O_DIM + lane + j * 32];
            x[j] = v; s += v; ss += v * v;
        }
        #pragma unroll
        for (int off = 16; off; off >>= 1) {
            s  += __shfl_xor(s,  off, 32);
            ss += __shfl_xor(ss, off, 32);
        }
        const float mu   = s * (1.f / O_DIM);
        const float var  = ss * (1.f / O_DIM) - mu * mu;
        const float rstd = rsqrtf(var + LN_EPS);
        const int   cnt  = counts[n];
        #pragma unroll
        for (int j = 0; j < 8; ++j) {
            const int o  = lane + j * 32;
            float hl = (x[j] - mu) * rstd * ln_w[o] + ln_b[o];
            float g  = 0.5f * hl * (1.f + erff(hl * 0.70710678118f));
            out[(size_t)n * O_DIM + o] = (cnt > 0) ? g : no_news[o];
        }
    }
}

// ---------------------------------------------------------------------------
extern "C" void kernel_launch(void* const* d_in, const int* in_sizes, int n_in,
                              void* d_out, int out_size, void* d_ws, size_t ws_size,
                              hipStream_t stream) {
    (void)in_sizes; (void)n_in; (void)out_size; (void)ws_size;

    const float* articles = (const float*)d_in[0];
    const int*   counts   = (const int*)  d_in[1];
    const float* attn_w   = (const float*)d_in[2];
    const float* attn_b   = (const float*)d_in[3];
    const float* proj_w   = (const float*)d_in[4];
    const float* proj_b   = (const float*)d_in[5];
    const float* ln_w     = (const float*)d_in[6];
    const float* ln_b     = (const float*)d_in[7];
    const float* no_news  = (const float*)d_in[8];
    float* out = (float*)d_out;

    __bf16* poolBf = (__bf16*)d_ws;                              // 8192*768 bf16
    __bf16* bw     = poolBf + (size_t)N_ROWS * D_DIM;            // 768*256 bf16

    pack_projw_kernel<<<(D_DIM * O_DIM) / 256, 256, 0, stream>>>(proj_w, bw);
    attn_pool_kernel<<<N_ROWS, 256, 0, stream>>>(articles, counts, attn_w,
                                                 attn_b, poolBf);
    gemm_ln_gelu_kernel<<<N_ROWS / 16, 256, 0, stream>>>(poolBf, bw, proj_b,
                                                         ln_w, ln_b, counts,
                                                         no_news, out);
}
